// QMixCriticMixer_47201690583219
// MI455X (gfx1250) — compile-verified, matrix operand order checked by reading, not run on hardware
//
#include <hip/hip_runtime.h>
#include <hip/hip_bf16.h>
#include <math.h>

// QMIX critic mixer: B=32, T=128, A=16, ACT=20, S=512, E=64
// N = 4096 samples, M = A*ACT = 320 counterfactual rows per sample.

typedef __attribute__((ext_vector_type(16))) __bf16 v16bf;
typedef __attribute__((ext_vector_type(8)))  float  v8f;
typedef float f4v __attribute__((ext_vector_type(4)));

// Explicit global-address-space pointers so vmem ops lower to global_* (LOADcnt
// only) instead of flat_* (LOADcnt+DScnt).
typedef const f4v __attribute__((address_space(1)))* gld4;
typedef       f4v __attribute__((address_space(1)))* gst4;

#define NS      4096
#define SDIM    512
#define ADIM    16
#define ACTN    20
#define EDIM    64
#define MDIM    320          // A*ACT
#define ST_TILES 256         // 4096/16 sample tiles
#define FB_TILES 19          // 1216/64 feature blocks: [0,16)=w1, 16=b1, 17=wf, 18=v1

#if defined(__has_builtin)
#if __has_builtin(__builtin_amdgcn_tensor_load_to_lds) && __has_builtin(__builtin_amdgcn_s_wait_tensorcnt)
#define HAVE_TDM 1
#endif
#endif
#ifndef HAVE_TDM
#define HAVE_TDM 0
#endif

#if HAVE_TDM
typedef unsigned int v4u __attribute__((ext_vector_type(4)));
typedef int          v4i __attribute__((ext_vector_type(4)));
typedef int          v8i __attribute__((ext_vector_type(8)));
#endif

// Load 16 fp32 in the 16-bit-A/B WMMA lane striping (elems 0..7 -> p[0..7],
// elems 8..15 -> p[16..23]) and pack to bf16. Lowers to 4x global_load_b128
// + 8x v_cvt_pk_bf16_f32.
static __device__ inline v16bf load_cvt16(const float* __restrict__ p) {
    gld4 q = (gld4)(const f4v*)p;
    f4v x0 = q[0], x1 = q[1], x2 = q[4], x3 = q[5];
    v16bf r;
    r[0]=(__bf16)x0.x;  r[1]=(__bf16)x0.y;  r[2]=(__bf16)x0.z;  r[3]=(__bf16)x0.w;
    r[4]=(__bf16)x1.x;  r[5]=(__bf16)x1.y;  r[6]=(__bf16)x1.z;  r[7]=(__bf16)x1.w;
    r[8]=(__bf16)x2.x;  r[9]=(__bf16)x2.y;  r[10]=(__bf16)x2.z; r[11]=(__bf16)x2.w;
    r[12]=(__bf16)x3.x; r[13]=(__bf16)x3.y; r[14]=(__bf16)x3.z; r[15]=(__bf16)x3.w;
    return r;
}

// ---------------------------------------------------------------------------
// Kernel 1: fused hypernetwork GEMM  Y[N,1216] = state[N,512] @ Wcat^T
// One wave per 16x64 output tile (4 accumulators share one A operand).
// Region select is wave-uniform. w1 is stored TRANSPOSED per sample ([e][a],
// bf16) so kernel 2's B operand becomes one contiguous 16B LDS read per lane.
// ---------------------------------------------------------------------------
__global__ __launch_bounds__(128) void qmix_hyper_gemm(
    const float* __restrict__ state,
    const float* __restrict__ Wh1, const float* __restrict__ bh1,
    const float* __restrict__ Wb1, const float* __restrict__ bb1,
    const float* __restrict__ Whf, const float* __restrict__ bhf,
    const float* __restrict__ Wv1, const float* __restrict__ bv1,
    __bf16* __restrict__ w1t, float* __restrict__ b1o,
    float*  __restrict__ wfo, float* __restrict__ vho)
{
    const int lane = threadIdx.x & 31;
    const int wg   = blockIdx.x * 4 + (threadIdx.x >> 5);
    const int st   = wg / FB_TILES;     // sample tile (GEMM M)
    const int fb   = wg % FB_TILES;     // 64-wide feature block (GEMM N / 4 tiles)

    const int half = lane >> 4;
    const int l15  = lane & 15;
    const int lk   = half * 8;

    const float* arow = state + (size_t)(st * 16 + l15) * SDIM;

    // Per-lane weight-row pointers + biases for the 4 column tiles.
    const float* wrow[4];
    float bias[4];
    if (fb < 16) {
        #pragma unroll
        for (int et = 0; et < 4; ++et) {
            const int fg = fb * 64 + et * 16 + l15;
            wrow[et] = Wh1 + (size_t)fg * SDIM;  bias[et] = bh1[fg];
        }
    } else if (fb == 16) {
        #pragma unroll
        for (int et = 0; et < 4; ++et) {
            const int fl = et * 16 + l15;
            wrow[et] = Wb1 + (size_t)fl * SDIM;  bias[et] = bb1[fl];
        }
    } else if (fb == 17) {
        #pragma unroll
        for (int et = 0; et < 4; ++et) {
            const int fl = et * 16 + l15;
            wrow[et] = Whf + (size_t)fl * SDIM;  bias[et] = bhf[fl];
        }
    } else {
        #pragma unroll
        for (int et = 0; et < 4; ++et) {
            const int fl = et * 16 + l15;
            wrow[et] = Wv1 + (size_t)fl * SDIM;  bias[et] = bv1[fl];
        }
    }

    v8f c[4] = {v8f{}, v8f{}, v8f{}, v8f{}};
    for (int k0 = 0; k0 < SDIM; k0 += 32) {
        const v16bf a = load_cvt16(arow + k0 + lk);   // shared by 4 WMMAs
        #pragma unroll
        for (int et = 0; et < 4; ++et) {
            const v16bf b = load_cvt16(wrow[et] + k0 + lk);
            c[et] = __builtin_amdgcn_wmma_f32_16x16x32_bf16(
                        false, a, false, b, (short)0, c[et], false, false);
        }
    }

    // Epilogue: lane holds column (l15 of tile et), rows r + half*8.
    if (fb < 16) {
        // w1 region: fg = fb*64 + fl -> a-index = fb, e-index = fl.
        // Transposed per-sample index: fl*16 + fb.
        #pragma unroll
        for (int et = 0; et < 4; ++et) {
            const int fl = et * 16 + l15;
            #pragma unroll
            for (int r = 0; r < 8; ++r) {
                const int samp = st * 16 + half * 8 + r;
                w1t[(size_t)samp * 1024 + fl * 16 + fb] = (__bf16)(c[et][r] + bias[et]);
            }
        }
    } else {
        float* dst = (fb == 16) ? b1o : (fb == 17) ? wfo : vho;
        const bool relu = (fb == 18);
        #pragma unroll
        for (int et = 0; et < 4; ++et) {
            const int fl = et * 16 + l15;
            #pragma unroll
            for (int r = 0; r < 8; ++r) {
                const int samp = st * 16 + half * 8 + r;
                float val = c[et][r] + bias[et];
                if (relu) val = fmaxf(val, 0.0f);
                dst[(size_t)samp * 64 + fl] = val;
            }
        }
    }
}

// ---------------------------------------------------------------------------
// Kernel 2: per-sample counterfactual mix. One 640-thread block (20 waves)
// per sample; wave w owns M-tile w (16 rows of the 320), computing all 4
// E-tiles via bf16 WMMA (K=16 padded to 32), then fused ELU + wf-dot + V.
// The 2KB w1 panel is staged into LDS by the Tensor Data Mover (one wave
// issues tensor_load_to_lds, waits TENSORcnt, barrier publishes it).
// ---------------------------------------------------------------------------
__global__ __launch_bounds__(640) void qmix_mix(
    const float* __restrict__ q_agents, const int* __restrict__ actions,
    const __bf16* __restrict__ w1t, const float* __restrict__ b1o,
    const float* __restrict__ wfo, const float* __restrict__ vho,
    const float* __restrict__ Wv2, const float* __restrict__ bv2,
    float* __restrict__ out)
{
    __shared__ float qa[ADIM * ACTN];    // q_agents row: [16,20]
    __shared__ float qts[ADIM];          // per-agent taken-action q
    __shared__ float b1s[EDIM], wfs[EDIM];
    __shared__ float red[EDIM];
    __shared__ float vtot;
    __shared__ unsigned int w1su[512];   // 1024 bf16, transposed [e][a]

    const int n   = blockIdx.x;
    const int tid = threadIdx.x;

    if (tid < MDIM) qa[tid] = q_agents[(size_t)n * MDIM + tid];
    if (tid < EDIM) {
        b1s[tid] = b1o[(size_t)n * EDIM + tid];
        wfs[tid] = wfo[(size_t)n * EDIM + tid];
        red[tid] = vho[(size_t)n * EDIM + tid] * Wv2[tid];
    }

#if HAVE_TDM
    if (tid < 32) {
        // D# for a 1024-element (2-byte) contiguous tile: global -> LDS.
        const unsigned long long ga =
            (unsigned long long)(const void*)(w1t + (size_t)n * 1024);
        const unsigned lds_off =
            (unsigned)(unsigned long long)(__attribute__((address_space(3))) unsigned int*)w1su;
        v4u g0;
        g0.x = 1u;                                   // count=1, user mode
        g0.y = lds_off;                              // lds_addr (bytes)
        g0.z = (unsigned)(ga & 0xFFFFFFFFu);         // global_addr[31:0]
        g0.w = (unsigned)((ga >> 32) & 0x01FFFFFFu)  // global_addr[56:32]
             | (2u << 30);                           // type=2 (image)
        v8i g1;
        g1[0] = (1 << 16);          // wg_mask=0, data_size=1 (2 bytes)
        g1[1] = (1024 << 16);       // tensor_dim0[15:0] in bits 63:48
        g1[2] = (1 << 16);          // tensor_dim0 hi=0 | tensor_dim1=1
        g1[3] = (1024 << 16);       // tensor_dim1 hi=0 | tile_dim0=1024
        g1[4] = 1;                  // tile_dim1=1, tile_dim2=0
        g1[5] = 1024;               // tensor_dim0_stride[31:0]
        g1[6] = 0;                  // stride0 hi | stride1 lo
        g1[7] = 0;                  // stride1 hi
        v4i gz = {0, 0, 0, 0};
#if __clang_major__ >= 23
        v8i gz8 = {0, 0, 0, 0, 0, 0, 0, 0};
        __builtin_amdgcn_tensor_load_to_lds(g0, g1, gz, gz, gz8, 0);
#else
        __builtin_amdgcn_tensor_load_to_lds(g0, g1, gz, gz, 0);
#endif
        __builtin_amdgcn_s_wait_tensorcnt(0);
    }
#else
    if (tid < 512)  // fallback: coalesced b32 stage of the bf16 w1 panel
        w1su[tid] = ((const unsigned int*)(w1t + (size_t)n * 1024))[tid];
#endif
    __syncthreads();

    if (tid < ADIM) {
        const int act = actions[n * ADIM + tid];
        qts[tid] = qa[tid * ACTN + act];
    }
    if (tid == 0) {
        float s = bv2[0];
        for (int e = 0; e < EDIM; ++e) s += red[e];
        vtot = s;
    }
    __syncthreads();

    const int wid  = tid >> 5;          // M-tile 0..19
    const int lane = tid & 31;
    const int half = lane >> 4;
    const int l15  = lane & 15;
    const int lk   = half * 8;
    const __bf16* w1s = (const __bf16*)w1su;

    // Counterfactual A row (row m, K = agent index 0..15, padded to 32).
    const int m  = wid * 16 + l15;
    const int ia = m / ACTN;
    const int ac = m % ACTN;
    const float diag = qa[ia * ACTN + ac];

    v16bf a = {};
    #pragma unroll
    for (int j = 0; j < 8; ++j) {       // elems 0..7 = K lk..lk+7; 8..15 stay 0
        const int k = lk + j;
        a[j] = (__bf16)((k == ia) ? diag : qts[k]);
    }

    v8f acc[4];
    #pragma unroll
    for (int et = 0; et < 4; ++et) {
        // One aligned 16B LDS read: w1t[e = et*16+l15][a = lk..lk+7].
        union { uint4 u; __bf16 h[8]; } uu;
        uu.u = *(const uint4*)(w1s + ((et * 16 + l15) * 16 + lk));
        v16bf b = {};
        #pragma unroll
        for (int j = 0; j < 8; ++j) b[j] = uu.h[j];
        v8f z = {};
        acc[et] = __builtin_amdgcn_wmma_f32_16x16x32_bf16(
                      false, a, false, b, (short)0, z, false, false);
    }

    // Fused epilogue: hidden = elu(acc + b1), p = hidden . wf  (lane = col e).
    float p[8];
    #pragma unroll
    for (int r = 0; r < 8; ++r) p[r] = 0.0f;
    #pragma unroll
    for (int et = 0; et < 4; ++et) {
        const float bb = b1s[et * 16 + l15];
        const float wf = wfs[et * 16 + l15];
        #pragma unroll
        for (int r = 0; r < 8; ++r) {
            const float x = acc[et][r] + bb;
            const float h = (x > 0.0f) ? x : (__expf(x) - 1.0f);
            p[r] += h * wf;
        }
    }
    // Reduce across the 16 lanes holding the same rows (masks stay in-half).
    #pragma unroll
    for (int mask = 1; mask <= 8; mask <<= 1)
        #pragma unroll
        for (int r = 0; r < 8; ++r)
            p[r] += __shfl_xor(p[r], mask, 32);

    if (l15 == 0) {
        const float v = vtot;
        f4v lo = { p[0] + v, p[1] + v, p[2] + v, p[3] + v };
        f4v hi = { p[4] + v, p[5] + v, p[6] + v, p[7] + v };
        gst4 o = (gst4)(f4v*)(out + (size_t)n * MDIM + wid * 16 + half * 8);
        o[0] = lo;
        o[1] = hi;
    }
}

// ---------------------------------------------------------------------------
// Launch
// ---------------------------------------------------------------------------
extern "C" void kernel_launch(void* const* d_in, const int* in_sizes, int n_in,
                              void* d_out, int out_size, void* d_ws, size_t ws_size,
                              hipStream_t stream) {
    (void)in_sizes; (void)n_in; (void)out_size; (void)ws_size;
    const float* q_agents = (const float*)d_in[0];
    const int*   actions  = (const int*)  d_in[1];
    const float* state    = (const float*)d_in[2];
    const float* Wh1 = (const float*)d_in[3];  const float* bh1 = (const float*)d_in[4];
    const float* Wb1 = (const float*)d_in[5];  const float* bb1 = (const float*)d_in[6];
    const float* Whf = (const float*)d_in[7];  const float* bhf = (const float*)d_in[8];
    const float* Wv1 = (const float*)d_in[9];  const float* bv1 = (const float*)d_in[10];
    const float* Wv2 = (const float*)d_in[11]; const float* bv2 = (const float*)d_in[12];
    float* out = (float*)d_out;

    // Workspace: w1 transposed (bf16, 8MB) | b1 (1MB) | wf (1MB) | v-hidden (1MB)
    __bf16* w1t = (__bf16*)d_ws;
    float*  b1o = (float*)((char*)d_ws + (size_t)NS * 1024 * sizeof(__bf16));
    float*  wfo = b1o + (size_t)NS * EDIM;
    float*  vho = wfo + (size_t)NS * EDIM;

    // Kernel 1: 256*19 = 4864 waves (16x64 tile each), 4 waves/block.
    qmix_hyper_gemm<<<(ST_TILES * FB_TILES) / 4, 128, 0, stream>>>(
        state, Wh1, bh1, Wb1, bb1, Whf, bhf, Wv1, bv1, w1t, b1o, wfo, vho);

    // Kernel 2: one block per sample.
    qmix_mix<<<NS, 640, 0, stream>>>(
        q_agents, actions, w1t, b1o, wfo, vho, Wv2, bv2, out);
}
